// MultiHeadAttention_20650202759203
// MI455X (gfx1250) — compile-verified, matrix operand order checked by reading, not run on hardware
//
#include <hip/hip_runtime.h>

// ---------------------------------------------------------------------------
// MI455X (gfx1250) multi-head attention forward.
// - All GEMM-shaped math on V_WMMA_F32_16X16X32_BF16 (wave32).
// - LDS tile staging via GLOBAL_LOAD_ASYNC_TO_LDS_B128 (ASYNCcnt-tracked DMA,
//   no VGPR round-trip), double-buffered in the projection GEMMs.
// ---------------------------------------------------------------------------

typedef __attribute__((ext_vector_type(16))) __bf16 v16bf;
typedef __attribute__((ext_vector_type(8)))  float  v8f;

union Frag32 {           // one 16x32 bf16 WMMA operand = 8 VGPRs = 32 bytes
    uint4 q[2];
    v16bf v;
};

__device__ __forceinline__ unsigned short f2bf(float f) {
    unsigned u = __builtin_bit_cast(unsigned, f);
    u += 0x7fffu + ((u >> 16) & 1u);          // round-to-nearest-even
    return (unsigned short)(u >> 16);
}

__device__ __forceinline__ v8f v8f_zero() {
    v8f z;
#pragma unroll
    for (int i = 0; i < 8; ++i) z[i] = 0.0f;
    return z;
}

__device__ __forceinline__ v8f wmma_bf16(v16bf a, v16bf b, v8f c) {
    return __builtin_amdgcn_wmma_f32_16x16x32_bf16(
        false, a, false, b, (short)0, c, false, false);
}

// 16-byte async DMA: global -> LDS, tracked by ASYNCcnt (no VGPR data path).
// VDST operand = 32-bit LDS byte address (low 32 bits of the flat pointer).
__device__ __forceinline__ void async_cp16(const void* g, void* l) {
    unsigned lds = (unsigned)(unsigned long long)l;
    asm volatile("global_load_async_to_lds_b128 %0, %1, off"
                 :: "v"(lds), "v"(g) : "memory");
}
__device__ __forceinline__ void wait_async0() {
    asm volatile("s_wait_asynccnt 0" ::: "memory");
}
__device__ __forceinline__ void wait_async4() {
    asm volatile("s_wait_asynccnt 4" ::: "memory");
}

// ---------------------------------------------------------------------------
// fp32 -> bf16 bulk conversion (n must be a multiple of 1024; it is here)
// ---------------------------------------------------------------------------
__global__ __launch_bounds__(256) void cvt_f32_bf16(
    const float* __restrict__ src, unsigned short* __restrict__ dst, int n) {
    int i = (blockIdx.x * 256 + threadIdx.x) * 4;
    if (i + 3 < n) {
        float4 v = *(const float4*)(src + i);
        uint2 o;
        o.x = (unsigned)f2bf(v.x) | ((unsigned)f2bf(v.y) << 16);
        o.y = (unsigned)f2bf(v.z) | ((unsigned)f2bf(v.w) << 16);
        *(uint2*)(dst + i) = o;
    }
}

// ---------------------------------------------------------------------------
// C[M,N] = A[M,K] @ B[N,K]^T + bias[N]   (torch Linear: x @ W.T + b)
// A, B bf16; accumulate f32; output bf16 (OUT_F32=0) or f32 (OUT_F32=1).
// Block tile 128x128, 256 threads = 8 waves (4 M x 2 N), wave tile 32x64.
// Double-buffered LDS tiles filled by async global->LDS DMA.
// ---------------------------------------------------------------------------
template <int OUT_F32>
__global__ __launch_bounds__(256) void gemm_bf16_nt(
    const unsigned short* __restrict__ A, const unsigned short* __restrict__ B,
    const float* __restrict__ bias, void* __restrict__ Cout,
    int M, int N, int K) {
    __shared__ __align__(16) unsigned short As[2][128 * 32];
    __shared__ __align__(16) unsigned short Bs[2][128 * 32];

    const int tid  = threadIdx.x;
    const int lane = tid & 31, wave = tid >> 5;
    const int ln = lane & 15, hh = lane >> 4;
    const int wm = wave >> 1, wn = wave & 1;
    const int m0 = blockIdx.y * 128, n0 = blockIdx.x * 128;

    // this thread's 2 chunks (16B each) of the A tile and of the B tile
    const int c0row = (tid + 0)   >> 2, c0seg = (tid + 0)   & 3;
    const int c1row = (tid + 256) >> 2, c1seg = (tid + 256) & 3;

    auto stage = [&](int k0, int buf) {
        async_cp16(&A[(size_t)(m0 + c0row) * K + k0 + c0seg * 8],
                   &As[buf][c0row * 32 + c0seg * 8]);
        async_cp16(&B[(size_t)(n0 + c0row) * K + k0 + c0seg * 8],
                   &Bs[buf][c0row * 32 + c0seg * 8]);
        async_cp16(&A[(size_t)(m0 + c1row) * K + k0 + c1seg * 8],
                   &As[buf][c1row * 32 + c1seg * 8]);
        async_cp16(&B[(size_t)(n0 + c1row) * K + k0 + c1seg * 8],
                   &Bs[buf][c1row * 32 + c1seg * 8]);
    };

    v8f acc[2][4];
#pragma unroll
    for (int mt = 0; mt < 2; ++mt)
#pragma unroll
        for (int nt = 0; nt < 4; ++nt) acc[mt][nt] = v8f_zero();

    const int nT = K / 32;
    stage(0, 0);
    for (int t = 0; t < nT; ++t) {
        const int buf = t & 1;
        __syncthreads();                 // everyone done reading buf^1 (iter t-1)
        if (t + 1 < nT) {
            stage((t + 1) * 32, buf ^ 1);
            wait_async4();               // newest 4 = tile t+1; older (tile t) done
        } else {
            wait_async0();
        }
        __syncthreads();                 // tile t visible to all waves

        Frag32 aF[2], bF[4];
#pragma unroll
        for (int mt = 0; mt < 2; ++mt) {
            int r = wm * 32 + mt * 16 + ln;
            aF[mt].q[0] = *(const uint4*)&As[buf][r * 32 + 8 * hh];
            aF[mt].q[1] = *(const uint4*)&As[buf][r * 32 + 16 + 8 * hh];
        }
#pragma unroll
        for (int nt = 0; nt < 4; ++nt) {
            int r = wn * 64 + nt * 16 + ln;
            bF[nt].q[0] = *(const uint4*)&Bs[buf][r * 32 + 8 * hh];
            bF[nt].q[1] = *(const uint4*)&Bs[buf][r * 32 + 16 + 8 * hh];
        }
#pragma unroll
        for (int mt = 0; mt < 2; ++mt)
#pragma unroll
            for (int nt = 0; nt < 4; ++nt)
                acc[mt][nt] = wmma_bf16(aF[mt].v, bF[nt].v, acc[mt][nt]);
    }

    // epilogue: C/D layout is lane = N, vgpr r -> M = r + 8*(lane>>4)
#pragma unroll
    for (int nt = 0; nt < 4; ++nt) {
        int gcol = n0 + wn * 64 + nt * 16 + ln;
        float bvv = bias[gcol];
#pragma unroll
        for (int mt = 0; mt < 2; ++mt) {
#pragma unroll
            for (int r = 0; r < 8; ++r) {
                int grow = m0 + wm * 32 + mt * 16 + 8 * hh + r;
                float v = acc[mt][nt][r] + bvv;
                if (OUT_F32)
                    ((float*)Cout)[(size_t)grow * N + gcol] = v;
                else
                    ((unsigned short*)Cout)[(size_t)grow * N + gcol] = f2bf(v);
            }
        }
    }
}

// ---------------------------------------------------------------------------
// Flash attention: one block = (batch, head, 128-query tile), 4 waves.
// Each wave owns 32 queries; streams 2048 keys in 64-key LDS tiles.
// K tile arrives via async global->LDS DMA; V is transposed through VGPRs.
// ---------------------------------------------------------------------------
__global__ __launch_bounds__(128) void attn_kernel(
    const unsigned short* __restrict__ Qw, const unsigned short* __restrict__ Kw,
    const unsigned short* __restrict__ Vw, unsigned short* __restrict__ Cw) {
    __shared__ __align__(16) unsigned short Kt[64 * 64];   // [key][dk]
    __shared__ __align__(16) unsigned short Vt[64 * 64];   // [dk][key] (transposed)
    __shared__ __align__(16) unsigned short Pt[128 * 64];  // [query][key] probs

    const int tid  = threadIdx.x;
    const int lane = tid & 31, wave = tid >> 5;
    const int ln = lane & 15, hh = lane >> 4;

    const int idx = blockIdx.x;
    const int b  = idx >> 8;
    const int h  = (idx >> 4) & 15;
    const int qb = idx & 15;

    const size_t tok0 = (size_t)b * 2048 + (size_t)qb * 128;
    const unsigned short* Qg = Qw + tok0 * 1024 + h * 64;
    const unsigned short* Kg = Kw + (size_t)b * 2048 * 1024 + h * 64;
    const unsigned short* Vg = Vw + (size_t)b * 2048 * 1024 + h * 64;
    unsigned short*       Cg = Cw + tok0 * 1024 + h * 64;

    // Q fragments live in registers for the whole kernel: 2 M-tiles x 2 K-tiles
    Frag32 aQ[2][2];
#pragma unroll
    for (int mt = 0; mt < 2; ++mt) {
        int row = wave * 32 + mt * 16 + ln;
#pragma unroll
        for (int kt = 0; kt < 2; ++kt) {
            aQ[mt][kt].q[0] = *(const uint4*)&Qg[(size_t)row * 1024 + kt * 32 + 8 * hh];
            aQ[mt][kt].q[1] = *(const uint4*)&Qg[(size_t)row * 1024 + kt * 32 + 16 + 8 * hh];
        }
    }

    v8f O[2][4];
#pragma unroll
    for (int mt = 0; mt < 2; ++mt)
#pragma unroll
        for (int nd = 0; nd < 4; ++nd) O[mt][nd] = v8f_zero();

    const float NEG_INF = -__builtin_inff();
    float mrow[2][8], lrow[2][8];
#pragma unroll
    for (int mt = 0; mt < 2; ++mt)
#pragma unroll
        for (int r = 0; r < 8; ++r) { mrow[mt][r] = NEG_INF; lrow[mt][r] = 0.0f; }

    const float c1 = 0.18033688011112042f;  // (1/sqrt(64)) * log2(e)

    for (int kb = 0; kb < 32; ++kb) {
        __syncthreads();
        // K tile: async DMA straight into LDS (512 x 16B chunks, 4 per thread)
#pragma unroll
        for (int i = 0; i < 4; ++i) {
            int c = tid + i * 128;
            int row = c >> 3, seg = c & 7;
            async_cp16(&Kg[(size_t)(kb * 64 + row) * 1024 + seg * 8],
                       &Kt[row * 64 + seg * 8]);
        }
        // V tile: load through VGPRs and store transposed
#pragma unroll
        for (int i = 0; i < 4; ++i) {
            int c = tid + i * 128;
            int row = c >> 3, seg = c & 7;
            uint4 vv = *(const uint4*)&Vg[(size_t)(kb * 64 + row) * 1024 + seg * 8];
            unsigned short e[8] = {
                (unsigned short)(vv.x), (unsigned short)(vv.x >> 16),
                (unsigned short)(vv.y), (unsigned short)(vv.y >> 16),
                (unsigned short)(vv.z), (unsigned short)(vv.z >> 16),
                (unsigned short)(vv.w), (unsigned short)(vv.w >> 16)};
#pragma unroll
            for (int j = 0; j < 8; ++j) Vt[(seg * 8 + j) * 64 + row] = e[j];
        }
        wait_async0();                    // K tile landed (V waits via barrier)
        __syncthreads();

        // S = Q @ K^T  : wave computes 32 queries x 64 keys
        v8f s[2][4];
#pragma unroll
        for (int mt = 0; mt < 2; ++mt)
#pragma unroll
            for (int nt = 0; nt < 4; ++nt) s[mt][nt] = v8f_zero();

#pragma unroll
        for (int nt = 0; nt < 4; ++nt) {
            int key = nt * 16 + ln;
#pragma unroll
            for (int kt = 0; kt < 2; ++kt) {
                Frag32 bK;
                bK.q[0] = *(const uint4*)&Kt[key * 64 + kt * 32 + 8 * hh];
                bK.q[1] = *(const uint4*)&Kt[key * 64 + kt * 32 + 16 + 8 * hh];
#pragma unroll
                for (int mt = 0; mt < 2; ++mt)
                    s[mt][nt] = wmma_bf16(aQ[mt][kt].v, bK.v, s[mt][nt]);
            }
        }

        // online softmax: each (half-wave, vgpr r) owns one full score row
#pragma unroll
        for (int mt = 0; mt < 2; ++mt) {
#pragma unroll
            for (int r = 0; r < 8; ++r) {
                float mx = fmaxf(fmaxf(s[mt][0][r], s[mt][1][r]),
                                 fmaxf(s[mt][2][r], s[mt][3][r]));
                mx = fmaxf(mx, __shfl_xor(mx, 1, 32));
                mx = fmaxf(mx, __shfl_xor(mx, 2, 32));
                mx = fmaxf(mx, __shfl_xor(mx, 4, 32));
                mx = fmaxf(mx, __shfl_xor(mx, 8, 32));
                mx *= c1;
                float mnew  = fmaxf(mrow[mt][r], mx);
                float alpha = __builtin_amdgcn_exp2f(mrow[mt][r] - mnew);
                mrow[mt][r] = mnew;

                int prow = wave * 32 + mt * 16 + 8 * hh + r;
                float rs = 0.0f;
#pragma unroll
                for (int nt = 0; nt < 4; ++nt) {
                    float p = __builtin_amdgcn_exp2f(s[mt][nt][r] * c1 - mnew);
                    rs += p;
                    Pt[prow * 64 + nt * 16 + ln] = f2bf(p);
                }
                rs += __shfl_xor(rs, 1, 32);
                rs += __shfl_xor(rs, 2, 32);
                rs += __shfl_xor(rs, 4, 32);
                rs += __shfl_xor(rs, 8, 32);
                lrow[mt][r] = lrow[mt][r] * alpha + rs;
#pragma unroll
                for (int nd = 0; nd < 4; ++nd) O[mt][nd][r] *= alpha;
            }
        }

        // O += P @ V  (P re-fragmented from own LDS region; DS is in-order/wave)
#pragma unroll
        for (int kt = 0; kt < 2; ++kt) {
            Frag32 aP[2];
#pragma unroll
            for (int mt = 0; mt < 2; ++mt) {
                int row = wave * 32 + mt * 16 + ln;
                aP[mt].q[0] = *(const uint4*)&Pt[row * 64 + kt * 32 + 8 * hh];
                aP[mt].q[1] = *(const uint4*)&Pt[row * 64 + kt * 32 + 16 + 8 * hh];
            }
#pragma unroll
            for (int nd = 0; nd < 4; ++nd) {
                int dk = nd * 16 + ln;
                Frag32 bV;
                bV.q[0] = *(const uint4*)&Vt[dk * 64 + kt * 32 + 8 * hh];
                bV.q[1] = *(const uint4*)&Vt[dk * 64 + kt * 32 + 16 + 8 * hh];
#pragma unroll
                for (int mt = 0; mt < 2; ++mt)
                    O[mt][nd] = wmma_bf16(aP[mt].v, bV.v, O[mt][nd]);
            }
        }
    }

    // finalize: context = O / l  -> bf16 workspace [token][1024]
#pragma unroll
    for (int mt = 0; mt < 2; ++mt) {
#pragma unroll
        for (int r = 0; r < 8; ++r) {
            float inv = 1.0f / lrow[mt][r];
            int row = wave * 32 + mt * 16 + 8 * hh + r;
#pragma unroll
            for (int nd = 0; nd < 4; ++nd)
                Cg[(size_t)row * 1024 + nd * 16 + ln] = f2bf(O[mt][nd][r] * inv);
        }
    }
}

// ---------------------------------------------------------------------------
// Host-side orchestration (all on `stream`, graph-capture safe)
// ---------------------------------------------------------------------------
extern "C" void kernel_launch(void* const* d_in, const int* in_sizes, int n_in,
                              void* d_out, int out_size, void* d_ws, size_t ws_size,
                              hipStream_t stream) {
    const float* x  = (const float*)d_in[0];
    const float* Wq = (const float*)d_in[1];
    const float* bq = (const float*)d_in[2];
    const float* Wk = (const float*)d_in[3];
    const float* bk = (const float*)d_in[4];
    const float* Wv = (const float*)d_in[5];
    const float* bv = (const float*)d_in[6];
    const float* Wo = (const float*)d_in[7];
    const float* bo = (const float*)d_in[8];
    float* out = (float*)d_out;

    // workspace layout (bf16 = unsigned short), total 88 MB
    char* ws = (char*)d_ws;
    unsigned short* xb  = (unsigned short*)ws;                 // 8192x1024
    unsigned short* Wqb = xb  + 8192u * 1024u;                 // 1024x1024 each
    unsigned short* Wkb = Wqb + 1024u * 1024u;
    unsigned short* Wvb = Wkb + 1024u * 1024u;
    unsigned short* Wob = Wvb + 1024u * 1024u;
    unsigned short* Qb  = Wob + 1024u * 1024u;                 // 8192x1024
    unsigned short* Kb  = Qb  + 8192u * 1024u;
    unsigned short* Vb  = Kb  + 8192u * 1024u;
    unsigned short* Cb  = Vb  + 8192u * 1024u;

    // 1) fp32 -> bf16
    cvt_f32_bf16<<<8192, 256, 0, stream>>>(x,  xb,  8192 * 1024);
    cvt_f32_bf16<<<1024, 256, 0, stream>>>(Wq, Wqb, 1024 * 1024);
    cvt_f32_bf16<<<1024, 256, 0, stream>>>(Wk, Wkb, 1024 * 1024);
    cvt_f32_bf16<<<1024, 256, 0, stream>>>(Wv, Wvb, 1024 * 1024);
    cvt_f32_bf16<<<1024, 256, 0, stream>>>(Wo, Wob, 1024 * 1024);

    // 2) Q/K/V projections: [8192,1024] @ [1024,1024]^T + bias -> bf16
    dim3 gg(1024 / 128, 8192 / 128);
    gemm_bf16_nt<0><<<gg, 256, 0, stream>>>(xb, Wqb, bq, Qb, 8192, 1024, 1024);
    gemm_bf16_nt<0><<<gg, 256, 0, stream>>>(xb, Wkb, bk, Kb, 8192, 1024, 1024);
    gemm_bf16_nt<0><<<gg, 256, 0, stream>>>(xb, Wvb, bv, Vb, 8192, 1024, 1024);

    // 3) flash attention: 4 batches x 16 heads x 16 query tiles
    attn_kernel<<<4 * 16 * 16, 128, 0, stream>>>(Qb, Kb, Vb, Cb);

    // 4) output projection -> fp32 d_out
    gemm_bf16_nt<1><<<gg, 256, 0, stream>>>(Cb, Wob, bo, out, 8192, 1024, 1024);
}
